// ThymeLM_71554155151765
// MI455X (gfx1250) — matrix operand order repaired and checked by reading, not compiled, not run on hardware
//
#include <hip/hip_runtime.h>
#include <math.h>

// Problem dims (match reference)
#define B_  4
#define S_  1024
#define V_  32000
#define E_  128
#define NA_ 12
#define NC_ 24
#define NS_ 576
#define NH_ 288          // NS/2
#define NHP_ 512         // NH padded to a multiple of 256 (kills in-loop guards)
#define CH_ 64           // scan chunks
#define CL_ 16           // steps per chunk (S_/CH_)
#define DECAY_ 0.9f
#define D16_ 0.18530201888518410f   // 0.9^16

// gfx1250 async global->LDS staging (ASYNCcnt path). Flip to 0 if asm rejects.
#define USE_ASYNC_LDS 1

typedef __attribute__((ext_vector_type(16))) __bf16       v16bf;
typedef __attribute__((ext_vector_type(8)))  float        v8f;
typedef __attribute__((ext_vector_type(8)))  unsigned int v8u;

__device__ __forceinline__ unsigned short f32_bf16(float f) {
    unsigned int u = __float_as_uint(f);
    u += 0x7FFFu + ((u >> 16) & 1u);       // round-to-nearest-even
    return (unsigned short)(u >> 16);
}

#if USE_ASYNC_LDS
__device__ __forceinline__ void async_copy_b128(unsigned lds_off, const void* gaddr) {
    // VDST = LDS byte address (low 32 bits of generic LDS pointer), VADDR = 64-bit global.
    asm volatile("global_load_async_to_lds_b128 %0, %1, off"
                 :: "v"(lds_off), "v"(gaddr) : "memory");
}
__device__ __forceinline__ void wait_asynccnt0() {
#if __has_builtin(__builtin_amdgcn_s_wait_asynccnt)
    __builtin_amdgcn_s_wait_asynccnt(0);
#else
    asm volatile("s_wait_asynccnt 0x0" ::: "memory");
#endif
}
#endif

// ---------------------------------------------------------------------------
// Kernel 1: per-token embedding gather + MLP (E->12 LN tanh -> 24 LN tanh).
// One wave32 per token; LayerNorms via lane shuffles.
// ---------------------------------------------------------------------------
__global__ __launch_bounds__(256) void tok_mlp_kernel(
    const int* __restrict__ ids, const float* __restrict__ emb,
    const float* __restrict__ W_ax, const float* __restrict__ b_ax,
    const float* __restrict__ g1, const float* __restrict__ be1,
    const float* __restrict__ W_comp, const float* __restrict__ b_comp,
    const float* __restrict__ g2, const float* __restrict__ be2,
    float* __restrict__ comp)
{
    const int token = blockIdx.x * 8 + (threadIdx.x >> 5);
    const int lane  = threadIdx.x & 31;
    const float* x = emb + (size_t)ids[token] * E_;

    float axv = 0.f;
    if (lane < NA_) {
        float s = 0.f;
        for (int e = 0; e < E_; ++e) s += x[e] * W_ax[e * NA_ + lane];
        axv = s + b_ax[lane];
    }
    float m = 0.f;
    for (int k = 0; k < NA_; ++k) m += __shfl(axv, k, 32);
    m *= (1.f / NA_);
    float d = axv - m, d2 = d * d;
    float v = 0.f;
    for (int k = 0; k < NA_; ++k) v += __shfl(d2, k, 32);
    v *= (1.f / NA_);
    float axn = 0.f;
    if (lane < NA_) axn = tanhf(d * rsqrtf(v + 1e-5f) * g1[lane] + be1[lane]);

    float s2 = 0.f;
    for (int i = 0; i < NA_; ++i) {
        float ai = __shfl(axn, i, 32);
        if (lane < NC_) s2 += ai * W_comp[i * NC_ + lane];
    }
    float cv = (lane < NC_) ? (s2 + b_comp[lane]) : 0.f;
    float m2 = 0.f;
    for (int k = 0; k < NC_; ++k) m2 += __shfl(cv, k, 32);
    m2 *= (1.f / NC_);
    float dd = cv - m2, dd2 = dd * dd;
    float v2 = 0.f;
    for (int k = 0; k < NC_; ++k) v2 += __shfl(dd2, k, 32);
    v2 *= (1.f / NC_);
    if (lane < NC_)
        comp[(size_t)token * NC_ + lane] =
            tanhf(dd * rsqrtf(v2 + 1e-5f) * g2[lane] + be2[lane]);
}

// ---------------------------------------------------------------------------
// Chunked linear-recurrence scan (see round-1 analysis).
// ---------------------------------------------------------------------------
__global__ void scan_partial_kernel(const float* __restrict__ comp,
                                    const float* __restrict__ mixw,
                                    float* __restrict__ carry)
{
    const int b = blockIdx.y, ch = blockIdx.x, t = threadIdx.x;
    const int i = t / NC_, j = t - i * NC_;
    const float mix = 1.f / (1.f + expf(-mixw[0]));
    __shared__ float cs[CL_ * NC_];
    const int base = (b * S_ + ch * CL_) * NC_;
    if (t < CL_ * NC_) cs[t] = comp[base + t];
    __syncthreads();
    float st = 0.f;
    for (int r = 0; r < CL_; ++r)
        st = DECAY_ * st + mix * cs[r * NC_ + i] * cs[r * NC_ + j];
    carry[(b * CH_ + ch) * NS_ + t] = st;
}

__global__ void scan_carry_kernel(float* __restrict__ carry,
                                  float* __restrict__ final_state)
{
    const int b = blockIdx.x, t = threadIdx.x;
    float run = 0.f;
    for (int ch = 0; ch < CH_; ++ch) {
        const int idx = (b * CH_ + ch) * NS_ + t;
        const float L = carry[idx];
        carry[idx] = run;              // carry INTO chunk ch
        run = L + D16_ * run;          // linear recurrence composition
    }
    final_state[b * NS_ + t] = run;
}

__global__ void scan_final_kernel(const float* __restrict__ comp,
                                  const float* __restrict__ mixw,
                                  const float* __restrict__ carry,
                                  unsigned short* __restrict__ flatbf)
{
    const int b = blockIdx.y, ch = blockIdx.x, t = threadIdx.x;
    const int i = t / NC_, j = t - i * NC_;
    const float mix = 1.f / (1.f + expf(-mixw[0]));
    __shared__ float cs[CL_ * NC_];
    const int base = (b * S_ + ch * CL_) * NC_;
    if (t < CL_ * NC_) cs[t] = comp[base + t];
    __syncthreads();
    float st = carry[(b * CH_ + ch) * NS_ + t];
    for (int r = 0; r < CL_; ++r) {
        st = DECAY_ * st + mix * cs[r * NC_ + i] * cs[r * NC_ + j];
        flatbf[(size_t)(b * S_ + ch * CL_ + r) * NS_ + t] = f32_bf16(st);
    }
}

// ---------------------------------------------------------------------------
// One-time weight prep: f32 [K][N] -> bf16 transposed+padded [Npad][K].
// Coalesced writes; strided reads are L2-covered (one-time cost).
// ---------------------------------------------------------------------------
__global__ void transpose_pad_bf16_kernel(const float* __restrict__ src,
                                          unsigned short* __restrict__ dst,
                                          int K, int N, int Npad)
{
    const int idx = blockIdx.x * 256 + threadIdx.x;
    if (idx >= Npad * K) return;
    const int n = idx / K, k = idx - n * K;
    dst[idx] = (n < N) ? f32_bf16(src[(size_t)k * N + n]) : (unsigned short)0;
}

// ---------------------------------------------------------------------------
// WMMA bf16 GEMM: C(MxNc) = A(MxK) * Bt(NbxK, pre-transposed) + bias.
// Block tile 64x256, 8 waves, each wave 32x64 -> 8 v_wmma per K-step.
// Double-buffered LDS (40KB); staging is pure contiguous b128 copies done
// with global_load_async_to_lds_b128 (ASYNCcnt), overlapped with WMMA work.
// EPI=0: +bias, exact GELU, bf16 out. EPI=1: +bias, f32 out.
// ---------------------------------------------------------------------------
template <int EPI>
__global__ __launch_bounds__(256) void gemm_bf16_wmma_kernel(
    const unsigned short* __restrict__ A,    // M x K (bf16 bits), K mult of 32
    const unsigned short* __restrict__ Bt,   // Nb x K (bf16 bits), Nb mult of 256
    const float* __restrict__ bias,          // Nc
    void* __restrict__ Cout,                 // M x Nc
    int M, int K, int Nc)
{
    __shared__ __align__(16) unsigned short As[2][64 * 32];   // [buf][m][k]
    __shared__ __align__(16) unsigned short Bs[2][256 * 32];  // [buf][n][k]

    const int tid  = threadIdx.x;
    const int lane = tid & 31;
    const int wave = tid >> 5;
    const int m0 = blockIdx.x * 64;
    const int n0 = blockIdx.y * 256;
    const int wm = (wave & 1) * 32;
    const int wn = (wave >> 1) * 64;
    const int l15 = lane & 15;
    const int hi  = lane >> 4;          // selects A K 0-7/16-23 vs 8-15/24-31; B likewise

    // --- per-thread staging addresses (hoisted; loop adds only +32 elems) ---
    const unsigned short* aSrc = A  + (size_t)(m0 + (tid >> 2)) * K + ((tid & 3) << 3);
    const unsigned short* bSrc = Bt + (size_t)(n0 + tid) * K;
    const int aOff = (tid >> 2) * 32 + ((tid & 3) << 3);
    const int bOff = tid * 32;
    unsigned short* aDstP[2] = { &As[0][aOff], &As[1][aOff] };
    unsigned short* bDstP[2] = { &Bs[0][bOff], &Bs[1][bOff] };
#if USE_ASYNC_LDS
    const unsigned aDst[2] = { (unsigned)(uintptr_t)aDstP[0], (unsigned)(uintptr_t)aDstP[1] };
    const unsigned bDst[2] = { (unsigned)(uintptr_t)bDstP[0], (unsigned)(uintptr_t)bDstP[1] };
#endif

    v8f acc[2][4] = {};
    const int nsteps = K >> 5;

    // --- prologue: stage tile 0 ---
#if USE_ASYNC_LDS
    async_copy_b128(aDst[0], aSrc);
    async_copy_b128(bDst[0],      bSrc);
    async_copy_b128(bDst[0] + 16, bSrc + 8);
    async_copy_b128(bDst[0] + 32, bSrc + 16);
    async_copy_b128(bDst[0] + 48, bSrc + 24);
    wait_asynccnt0();
#else
    {
        const uint4 ra  = *reinterpret_cast<const uint4*>(aSrc);
        const uint4 rb0 = *reinterpret_cast<const uint4*>(bSrc);
        const uint4 rb1 = *reinterpret_cast<const uint4*>(bSrc + 8);
        const uint4 rb2 = *reinterpret_cast<const uint4*>(bSrc + 16);
        const uint4 rb3 = *reinterpret_cast<const uint4*>(bSrc + 24);
        *reinterpret_cast<uint4*>(aDstP[0])      = ra;
        *reinterpret_cast<uint4*>(bDstP[0])      = rb0;
        *reinterpret_cast<uint4*>(bDstP[0] + 8)  = rb1;
        *reinterpret_cast<uint4*>(bDstP[0] + 16) = rb2;
        *reinterpret_cast<uint4*>(bDstP[0] + 24) = rb3;
    }
#endif
    __syncthreads();

    for (int it = 0; it < nsteps; ++it) {
        const int cur = it & 1;
        const int nxt = cur ^ 1;
        const bool have_next = (it + 1) < nsteps;
        const int kn = (it + 1) << 5;

        // issue next-tile staging first so it overlaps the WMMA work below
#if USE_ASYNC_LDS
        if (have_next) {
            async_copy_b128(aDst[nxt], aSrc + kn);
            async_copy_b128(bDst[nxt],      bSrc + kn);
            async_copy_b128(bDst[nxt] + 16, bSrc + kn + 8);
            async_copy_b128(bDst[nxt] + 32, bSrc + kn + 16);
            async_copy_b128(bDst[nxt] + 48, bSrc + kn + 24);
        }
#else
        uint4 ra{}, rb0{}, rb1{}, rb2{}, rb3{};
        if (have_next) {
            ra  = *reinterpret_cast<const uint4*>(aSrc + kn);
            rb0 = *reinterpret_cast<const uint4*>(bSrc + kn);
            rb1 = *reinterpret_cast<const uint4*>(bSrc + kn + 8);
            rb2 = *reinterpret_cast<const uint4*>(bSrc + kn + 16);
            rb3 = *reinterpret_cast<const uint4*>(bSrc + kn + 24);
        }
#endif

        // --- fragments from current buffers ---
        // A frag (ISA 16-bit A 16x32): lane<16 -> K 0-7 & 16-23, lane>=16 -> 8-15 & 24-31
        v16bf afrag[2];
        #pragma unroll
        for (int mi = 0; mi < 2; ++mi) {
            const uint4* arow = reinterpret_cast<const uint4*>(
                &As[cur][(wm + mi * 16 + l15) * 32]);
            const uint4 lo = arow[hi];
            const uint4 hi4 = arow[2 + hi];
            v8u au;
            au[0] = lo.x;  au[1] = lo.y;  au[2] = lo.z;  au[3] = lo.w;
            au[4] = hi4.x; au[5] = hi4.y; au[6] = hi4.z; au[7] = hi4.w;
            afrag[mi] = __builtin_bit_cast(v16bf, au);
        }
        // B frags + WMMA: N = lane&15, lane<16 -> K 0-7 & 16-23, lane>=16 -> 8-15 & 24-31
        #pragma unroll
        for (int t = 0; t < 4; ++t) {
            const uint4* brow = reinterpret_cast<const uint4*>(
                &Bs[cur][(wn + t * 16 + l15) * 32]);
            const uint4 blo = brow[hi];
            const uint4 bhi = brow[2 + hi];
            v8u bu;
            bu[0] = blo.x; bu[1] = blo.y; bu[2] = blo.z; bu[3] = blo.w;
            bu[4] = bhi.x; bu[5] = bhi.y; bu[6] = bhi.z; bu[7] = bhi.w;
            const v16bf bfv = __builtin_bit_cast(v16bf, bu);
            #pragma unroll
            for (int mi = 0; mi < 2; ++mi)
                acc[mi][t] = __builtin_amdgcn_wmma_f32_16x16x32_bf16(
                    false, afrag[mi], false, bfv, (short)0, acc[mi][t], false, false);
        }

#if USE_ASYNC_LDS
        if (have_next) wait_asynccnt0();
#else
        if (have_next) {
            *reinterpret_cast<uint4*>(aDstP[nxt])      = ra;
            *reinterpret_cast<uint4*>(bDstP[nxt])      = rb0;
            *reinterpret_cast<uint4*>(bDstP[nxt] + 8)  = rb1;
            *reinterpret_cast<uint4*>(bDstP[nxt] + 16) = rb2;
            *reinterpret_cast<uint4*>(bDstP[nxt] + 24) = rb3;
        }
#endif
        __syncthreads();
    }

    // Epilogue. C/D layout: VGPR r -> M = r + 8*(lane>=16), N = lane&15
    #pragma unroll
    for (int mi = 0; mi < 2; ++mi) {
        const int mbase = m0 + wm + mi * 16 + (hi << 3);
        #pragma unroll
        for (int t = 0; t < 4; ++t) {
            const int n = n0 + wn + t * 16 + l15;
            if (n < Nc) {
                const float bv = bias[n];
                #pragma unroll
                for (int r = 0; r < 8; ++r) {
                    const float xv = acc[mi][t][r] + bv;
                    const size_t off = (size_t)(mbase + r) * Nc + n;
                    if (EPI == 0) {   // exact GELU, bf16 out
                        const float gl = 0.5f * xv * (1.0f + erff(xv * 0.7071067811865475f));
                        reinterpret_cast<unsigned short*>(Cout)[off] = f32_bf16(gl);
                    } else {          // f32 logits out
                        reinterpret_cast<float*>(Cout)[off] = xv;
                    }
                }
            }
        }
    }
}

// ---------------------------------------------------------------------------
// Launcher
// ---------------------------------------------------------------------------
extern "C" void kernel_launch(void* const* d_in, const int* in_sizes, int n_in,
                              void* d_out, int out_size, void* d_ws, size_t ws_size,
                              hipStream_t stream)
{
    (void)in_sizes; (void)n_in; (void)out_size; (void)ws_size;

    const int*   input_ids = (const int*)  d_in[0];
    const float* emb       = (const float*)d_in[1];
    const float* W_ax      = (const float*)d_in[2];
    const float* b_ax      = (const float*)d_in[3];
    const float* g1        = (const float*)d_in[4];
    const float* be1       = (const float*)d_in[5];
    const float* W_comp    = (const float*)d_in[6];
    const float* b_comp    = (const float*)d_in[7];
    const float* g2        = (const float*)d_in[8];
    const float* be2       = (const float*)d_in[9];
    const float* mixw      = (const float*)d_in[10];
    const float* W1        = (const float*)d_in[11];
    const float* b1        = (const float*)d_in[12];
    const float* W2        = (const float*)d_in[13];
    const float* b2        = (const float*)d_in[14];

    // Workspace layout (16B aligned offsets)
    char* ws = (char*)d_ws;
    float*          comp   = (float*)         (ws + 0);        // 4096*24*4    = 393216
    float*          carry  = (float*)         (ws + 393216);   // 4*64*576*4   = 589824
    unsigned short* flatbf = (unsigned short*)(ws + 983040);   // 4096*576*2   = 4718592
    unsigned short* W1t    = (unsigned short*)(ws + 5701632);  // 512*576*2    = 589824
    unsigned short* hbf    = (unsigned short*)(ws + 6291456);  // 4096*288*2   = 2359296
    unsigned short* W2t    = (unsigned short*)(ws + 8650752);  // 32000*288*2  = 18432000

    float* logits = (float*)d_out;
    float* final_state = logits + (size_t)B_ * S_ * V_;

    tok_mlp_kernel<<<dim3((B_ * S_) / 8), 256, 0, stream>>>(
        input_ids, emb, W_ax, b_ax, g1, be1, W_comp, b_comp, g2, be2, comp);

    // One-time weight prep: W1 (576x288)->W1t (512x576), W2 (288x32000)->W2t (32000x288)
    transpose_pad_bf16_kernel<<<dim3((NHP_ * NS_ + 255) / 256), 256, 0, stream>>>(
        W1, W1t, NS_, NH_, NHP_);
    transpose_pad_bf16_kernel<<<dim3((V_ * NH_ + 255) / 256), 256, 0, stream>>>(
        W2, W2t, NH_, V_, V_);

    scan_partial_kernel<<<dim3(CH_, B_), NS_, 0, stream>>>(comp, mixw, carry);
    scan_carry_kernel<<<dim3(B_), NS_, 0, stream>>>(carry, final_state);
    scan_final_kernel<<<dim3(CH_, B_), NS_, 0, stream>>>(comp, mixw, carry, flatbf);

    // GEMM1: (4096x576)*(576x288) + b1, GELU -> bf16 h
    gemm_bf16_wmma_kernel<0><<<dim3((B_ * S_) / 64, NHP_ / 256), 256, 0, stream>>>(
        flatbf, W1t, b1, hbf, B_ * S_, NS_, NH_);
    // GEMM2: (4096x288)*(288x32000) + b2 -> f32 logits (HBM-store-bound stage)
    gemm_bf16_wmma_kernel<1><<<dim3((B_ * S_) / 64, V_ / 256), 256, 0, stream>>>(
        hbf, W2t, b2, logits, B_ * S_, V_, NH_);
}
